// MTRDecoder_17102559772673
// MI455X (gfx1250) — matrix operand order repaired and checked by reading, not compiled, not run on hardware
//
#include <hip/hip_runtime.h>

// ---------------- problem constants (match reference) ----------------
#define D_   256
#define H_   8
#define DH_  32
#define FF_  1024
#define NQ_  64
#define BS_  16
#define NK_  768
#define L_   128
#define NTOT_ (NQ_*BS_)

// ---------------- CDNA5 WMMA types ----------------
typedef __attribute__((ext_vector_type(16))) __bf16        v16bf;
typedef __attribute__((ext_vector_type(8)))  float         v8f;
typedef __attribute__((ext_vector_type(8)))  unsigned int  v8u;

// ---------------- CDNA5 async global->LDS copy (guarded) ----------------
#if defined(__has_builtin)
# if __has_builtin(__builtin_amdgcn_global_load_async_to_lds_b128)
#  define HAVE_ASYNC_LDS 1
# endif
# if __has_builtin(__builtin_amdgcn_s_wait_asynccnt)
#  define HAVE_WAIT_ASYNC 1
# endif
#endif

#ifdef HAVE_ASYNC_LDS
// Builtin signature (from compiler diagnostic): param is pointer to
// int __attribute__((vector_size(16))) in AS1 (global) / AS3 (LDS).
typedef int v4i_vs __attribute__((vector_size(16)));
typedef __attribute__((address_space(1))) v4i_vs* gptr128_t;
typedef __attribute__((address_space(3))) v4i_vs* lptr128_t;
#endif

// copy one 16-byte chunk global->LDS
__device__ __forceinline__ void copy16(unsigned short* dst, const unsigned short* src) {
#ifdef HAVE_ASYNC_LDS
  __builtin_amdgcn_global_load_async_to_lds_b128(
      (gptr128_t)(unsigned long long)src,
      (lptr128_t)(unsigned int)(unsigned long long)dst,
      0, 0);
#else
  *(uint4*)dst = *(const uint4*)src;
#endif
}

template <int N>
__device__ __forceinline__ void wait_async() {
#ifdef HAVE_ASYNC_LDS
# ifdef HAVE_WAIT_ASYNC
  __builtin_amdgcn_s_wait_asynccnt(N);
# else
  asm volatile("s_wait_asynccnt %0" :: "i"(N) : "memory");
# endif
#endif
}

__device__ __forceinline__ unsigned short f2bf(float f) {
  unsigned int u = __builtin_bit_cast(unsigned int, f);
  u += 0x7fffu + ((u >> 16) & 1u);          // round-to-nearest-even
  return (unsigned short)(u >> 16);
}

// Build a v16bf fragment from two contiguous 16-byte LDS chunks.
__device__ __forceinline__ v16bf frag2x16(const unsigned short* p0,
                                          const unsigned short* p1) {
  const uint4 a = *(const uint4*)p0;
  const uint4 b = *(const uint4*)p1;
  v8u u;
  u[0]=a.x; u[1]=a.y; u[2]=a.z; u[3]=a.w;
  u[4]=b.x; u[5]=b.y; u[6]=b.z; u[7]=b.w;
  return __builtin_bit_cast(v16bf, u);
}

// =====================================================================
// f32 -> packed bf16 (flat), n multiple of 4
// =====================================================================
__global__ __launch_bounds__(256) void cvt_bf16_kernel(
    const float* __restrict__ X, unsigned short* __restrict__ Y, int n4)
{
  const int i = blockIdx.x * 256 + threadIdx.x;
  if (i < n4) {
    const float4 v = ((const float4*)X)[i];
    uint2 u;
    u.x = (unsigned int)f2bf(v.x) | ((unsigned int)f2bf(v.y) << 16);
    u.y = (unsigned int)f2bf(v.z) | ((unsigned int)f2bf(v.w) << 16);
    ((uint2*)Y)[i] = u;
  }
}

// =====================================================================
// f32 W[K,N] -> bf16 Wt[N,K]  (32x32 LDS tile transpose)
// =====================================================================
__global__ __launch_bounds__(256) void cvt_bf16_t_kernel(
    const float* __restrict__ W, unsigned short* __restrict__ Wt, int K, int N)
{
  __shared__ unsigned short tile[32][33];
  const int n0 = blockIdx.x * 32, k0 = blockIdx.y * 32;
  const int tx = threadIdx.x & 31, ty = threadIdx.x >> 5;
  for (int i = 0; i < 4; ++i) {
    const int k = ty + i*8;
    tile[k][tx] = f2bf(W[(size_t)(k0 + k) * N + n0 + tx]);
  }
  __syncthreads();
  for (int i = 0; i < 4; ++i) {
    const int n = ty + i*8;
    Wt[(size_t)(n0 + n) * K + k0 + tx] = tile[tx][n];
  }
}

// row permutation [NQ,BS,D] -> bf16 [BS*NQ, D]
__global__ __launch_bounds__(256) void reorder_bf16_kernel(
    const float* __restrict__ X, unsigned short* __restrict__ Y)
{
  const int i = blockIdx.x * 256 + threadIdx.x;
  const int d = i & 255;
  const int n = i >> 8;
  const int b = n >> 6, q = n & 63;
  Y[i] = f2bf(X[(size_t)(q*BS_ + b)*D_ + d]);
}

// =====================================================================
// bf16-WMMA GEMM: out = act(A*Wt' [+ A2*W2t'] + bias [+ bias2])
// A: [M,K] bf16. Wt: [N,K] bf16 (pre-transposed). C f32 / Cbf bf16 optional.
// Block tile 128x64, 8 waves of 32x32, BK=64, double-buffered async
// global->LDS staging (ASYNCcnt pipeline). Padded LDS stride 72 keeps
// every fragment two contiguous conflict-free ds_load_b128.
// =====================================================================
__global__ __launch_bounds__(256) void gemm_bf16_kernel(
    const unsigned short* __restrict__ A,  const unsigned short* __restrict__ Wt,
    const float* __restrict__ bias,
    const unsigned short* __restrict__ A2, const unsigned short* __restrict__ W2t,
    const float* __restrict__ bias2,
    float* __restrict__ C, unsigned short* __restrict__ Cbf,
    int M, int N, int K, int relu)
{
  __shared__ unsigned short As[2][128*72];
  __shared__ unsigned short Bs[2][64*72];

  const int tid  = threadIdx.x;
  const int lane = tid & 31;
  const int wid  = tid >> 5;
  const int wm   = (wid & 3) * 32;
  const int wn   = (wid >> 2) * 32;
  const int bm   = blockIdx.x * 128;
  const int bn   = blockIdx.y * 64;

  const v8f zero = {0.f,0.f,0.f,0.f,0.f,0.f,0.f,0.f};
  v8f acc[2][2];
  for (int i = 0; i < 2; ++i) for (int j = 0; j < 2; ++j) acc[i][j] = zero;

  const int ka = (lane < 16) ? 0 : 8;   // A-frag k base
  const int kb = (lane < 16) ? 0 : 8;   // B-frag k base

  const unsigned short* APs[2] = {A, A2};
  const unsigned short* WPs[2] = {Wt, W2t};
  const int kst   = K >> 6;                       // 64-wide K stages per pass
  const int npass = (A2 != nullptr) ? 2 : 1;
  const int nst   = kst * npass;

  // Stage s: 6 x 16B async copies per thread (A:4, B:2).
  auto stage = [&](int buf, int s) {
    const unsigned short* Ap = APs[s / kst];
    const unsigned short* Wp = WPs[s / kst];
    const int kt = (s % kst) << 6;
    for (int u = tid; u < 1024; u += 256) {       // A: 128 rows x 8 chunks
      const int r = u >> 3, c = (u & 7) * 8;
      copy16(&As[buf][r*72 + c], &Ap[(size_t)(bm + r) * K + kt + c]);
    }
    for (int u = tid; u < 512; u += 256) {        // B: 64 cols x 8 chunks
      const int r = u >> 3, c = (u & 7) * 8;
      copy16(&Bs[buf][r*72 + c], &Wp[(size_t)(bn + r) * K + kt + c]);
    }
  };

  stage(0, 0);
  for (int s = 0; s < nst; ++s) {
    const int buf = s & 1;
    if (s + 1 < nst) { stage(buf ^ 1, s + 1); wait_async<6>(); }
    else             { wait_async<0>(); }
    __syncthreads();

    const int row = wm + (lane & 15);
    const int col = wn + (lane & 15);
    for (int ks = 0; ks < 64; ks += 32) {
      v16bf afr[2], bfr[2];
      afr[0] = frag2x16(&As[buf][ row     *72 + ks + ka], &As[buf][ row     *72 + ks + ka + 16]);
      afr[1] = frag2x16(&As[buf][(row+16)*72 + ks + ka], &As[buf][(row+16)*72 + ks + ka + 16]);
      bfr[0] = frag2x16(&Bs[buf][ col     *72 + ks + kb], &Bs[buf][ col     *72 + ks + kb + 16]);
      bfr[1] = frag2x16(&Bs[buf][(col+16)*72 + ks + kb], &Bs[buf][(col+16)*72 + ks + kb + 16]);
      for (int i = 0; i < 2; ++i)
        for (int j = 0; j < 2; ++j)
          acc[i][j] = __builtin_amdgcn_wmma_f32_16x16x32_bf16(
              false, afr[i], false, bfr[j], (short)0, acc[i][j], false, false);
    }
    __syncthreads();
  }

  const int n0 = bn + wn + (lane & 15);
  const int m0 = bm + wm + ((lane < 16) ? 0 : 8);
  for (int i = 0; i < 2; ++i) {
    for (int j = 0; j < 2; ++j) {
      const int n = n0 + 16*j;
      const float badd = (bias ? bias[n] : 0.f) + (bias2 ? bias2[n] : 0.f);
      for (int r = 0; r < 8; ++r) {
        const int m = m0 + 16*i + r;
        float v = acc[i][j][r] + badd;
        if (relu) v = fmaxf(v, 0.f);
        if (C)   C  [(size_t)m * N + n] = v;
        if (Cbf) Cbf[(size_t)m * N + n] = f2bf(v);
      }
    }
  }
}

// =====================================================================
// Dense self-attention, one block per (b,h); bf16 Q/K/V in, bf16 O out.
// Rows in [NQ,BS,D] layout (row = q*BS+b, cols h*32..h*32+31).
// =====================================================================
__global__ __launch_bounds__(128) void self_attn_kernel(
    const unsigned short* __restrict__ Qb, const unsigned short* __restrict__ Kb,
    const unsigned short* __restrict__ Vb, unsigned short* __restrict__ Ob)
{
  const int b = blockIdx.x >> 3;
  const int h = blockIdx.x & 7;
  const int t = threadIdx.x, lane = t & 31, w = t >> 5;

  __shared__ unsigned short Qs[64*40];   // [q][d]   stride 40
  __shared__ unsigned short Ks[64*40];   // [key][d] stride 40
  __shared__ unsigned short Vst[32*72];  // [d][key] stride 72 (transposed)
  __shared__ unsigned short Pb[64*72];   // [q][key] stride 72
  __shared__ float Sf[64*65];            // odd stride: conflict-free softmax

  // stage Q,K via async b128 copies: 64 rows x 4 chunks each
  for (int u = t; u < 256; u += 128) {
    const int q = u >> 2, c = (u & 3) * 8;
    const size_t src = (size_t)(q*BS_ + b)*D_ + h*DH_ + c;
    copy16(&Qs[q*40 + c], &Qb[src]);
    copy16(&Ks[q*40 + c], &Kb[src]);
  }
  // stage V transposed (gather; packed ds stores)
  for (int u = t; u < 512; u += 128) {
    const int d = u >> 4, kg = (u & 15) * 4;
    const unsigned int v0 = Vb[(size_t)((kg+0)*BS_ + b)*D_ + h*DH_ + d];
    const unsigned int v1 = Vb[(size_t)((kg+1)*BS_ + b)*D_ + h*DH_ + d];
    const unsigned int v2 = Vb[(size_t)((kg+2)*BS_ + b)*D_ + h*DH_ + d];
    const unsigned int v3 = Vb[(size_t)((kg+3)*BS_ + b)*D_ + h*DH_ + d];
    uint2 uu; uu.x = v0 | (v1 << 16); uu.y = v2 | (v3 << 16);
    *(uint2*)&Vst[d*72 + kg] = uu;
  }
  wait_async<0>();
  __syncthreads();

  const v8f zero = {0.f,0.f,0.f,0.f,0.f,0.f,0.f,0.f};
  const int ka = (lane < 16) ? 0 : 8;
  const int kb = (lane < 16) ? 0 : 8;

  // ---- scores ----
  {
    const int mt = 16*w;
    const int row = mt + (lane & 15);
    const v16bf af = frag2x16(&Qs[row*40 + ka], &Qs[row*40 + ka + 16]);
    for (int jt = 0; jt < 4; ++jt) {
      const int col = 16*jt + (lane & 15);
      const v16bf bf = frag2x16(&Ks[col*40 + kb], &Ks[col*40 + kb + 16]);
      const v8f s = __builtin_amdgcn_wmma_f32_16x16x32_bf16(
          false, af, false, bf, (short)0, zero, false, false);
      const int n  = 16*jt + (lane & 15);
      const int m0 = mt + ((lane < 16) ? 0 : 8);
      for (int r = 0; r < 8; ++r) Sf[(m0 + r)*65 + n] = s[r] * 0.17677669529663687f;
    }
  }
  __syncthreads();

  // ---- softmax per row, emit bf16 P ----
  if (t < 64) {
    float mx = -1e30f;
    for (int j = 0; j < 64; ++j) mx = fmaxf(mx, Sf[t*65 + j]);
    float s = 0.f;
    for (int j = 0; j < 64; ++j) { float e = __expf(Sf[t*65 + j] - mx); Sf[t*65 + j] = e; s += e; }
    const float inv = 1.f / s;
    for (int j = 0; j < 64; j += 2) {
      const unsigned int pk = (unsigned int)f2bf(Sf[t*65 + j] * inv)
                            | ((unsigned int)f2bf(Sf[t*65 + j + 1] * inv) << 16);
      *(unsigned int*)&Pb[t*72 + j] = pk;
    }
  }
  __syncthreads();

  // ---- O = P * V ----
  {
    const int mt = 16*w;
    v8f acc2[2] = {zero, zero};
    for (int ks = 0; ks < 64; ks += 32) {
      const int row = mt + (lane & 15);
      const v16bf af = frag2x16(&Pb[row*72 + ks + ka], &Pb[row*72 + ks + ka + 16]);
      for (int nt = 0; nt < 2; ++nt) {
        const int col = 16*nt + (lane & 15);
        const v16bf bf = frag2x16(&Vst[col*72 + ks + kb], &Vst[col*72 + ks + kb + 16]);
        acc2[nt] = __builtin_amdgcn_wmma_f32_16x16x32_bf16(
            false, af, false, bf, (short)0, acc2[nt], false, false);
      }
    }
    for (int nt = 0; nt < 2; ++nt) {
      const int n  = 16*nt + (lane & 15);
      const int m0 = mt + ((lane < 16) ? 0 : 8);
      for (int r = 0; r < 8; ++r) {
        const int q = m0 + r;
        Ob[(size_t)(q*BS_ + b)*D_ + h*DH_ + n] = f2bf(acc2[nt][r]);
      }
    }
  }
}

// =====================================================================
// Gathered local cross-attention (VALU; per-query gather precludes WMMA)
// Block per query n, wave per head. float4-vectorized key loads. bf16 out.
// =====================================================================
__global__ __launch_bounds__(256) void cross_attn_kernel(
    const float* __restrict__ CQ, const float* __restrict__ CQSE,
    const float* __restrict__ KC, const float* __restrict__ KP,
    const float* __restrict__ VM,
    const int* __restrict__ index_pair, const int* __restrict__ index_batch,
    const int* __restrict__ key_cnt,
    unsigned short* __restrict__ CO)
{
  const int n = blockIdx.x;
  const int t = threadIdx.x, lane = t & 31, h = t >> 5;

  __shared__ float qh[H_][64];
  __shared__ float ps[H_][L_];
  __shared__ int   gidx[L_];

  {
    const int d = t & 31;
    qh[h][d]      = CQ  [(size_t)n*D_ + t];
    qh[h][32 + d] = CQSE[(size_t)n*D_ + t];
  }
  const int b = index_batch[n];
  int off = 0;
  for (int i = 0; i < BS_; ++i) off += (i < b) ? key_cnt[i] : 0;
  if (t < L_) {
    const int idx = index_pair[(size_t)n*L_ + t];
    gidx[t] = (idx >= 0) ? (idx + off) : -1;
  }
  __syncthreads();

  for (int li = lane; li < L_; li += 32) {
    const int g = gidx[li];
    float s;
    if (g < 0) s = -1e9f;
    else {
      const float* kc = KC + (size_t)g*D_ + h*DH_;
      const float* kp = KP + (size_t)g*D_ + h*DH_;
      float a = 0.f;
      for (int d4 = 0; d4 < DH_; d4 += 4) {
        const float4 kc4 = *(const float4*)(kc + d4);
        const float4 kp4 = *(const float4*)(kp + d4);
        a += qh[h][d4+0]*(kc4.x+kp4.x) + qh[h][32+d4+0]*kp4.x;
        a += qh[h][d4+1]*(kc4.y+kp4.y) + qh[h][32+d4+1]*kp4.y;
        a += qh[h][d4+2]*(kc4.z+kp4.z) + qh[h][32+d4+2]*kp4.z;
        a += qh[h][d4+3]*(kc4.w+kp4.w) + qh[h][32+d4+3]*kp4.w;
      }
      s = a * 0.125f;
    }
    ps[h][li] = s;
  }
  __syncthreads();

  float m = -1e30f;
  for (int li = lane; li < L_; li += 32) m = fmaxf(m, ps[h][li]);
  for (int o = 16; o > 0; o >>= 1) m = fmaxf(m, __shfl_xor(m, o, 32));
  float sum = 0.f;
  for (int li = lane; li < L_; li += 32) {
    const float e = __expf(ps[h][li] - m);
    ps[h][li] = e; sum += e;
  }
  for (int o = 16; o > 0; o >>= 1) sum += __shfl_xor(sum, o, 32);
  const float inv = 1.f / sum;
  __syncthreads();

  const int d = lane;
  float oa = 0.f;
  for (int li = 0; li < L_; ++li) {
    const int g = gidx[li];
    if (g >= 0) oa += ps[h][li] * VM[(size_t)g*D_ + h*DH_ + d];
  }
  CO[(size_t)n*D_ + h*DH_ + d] = f2bf(oa * inv);
}

// =====================================================================
// y = LayerNorm(x + r); optional [NQ,BS]->[BS,NQ] input row permutation;
// optional packed-bf16 secondary output.
// =====================================================================
__global__ __launch_bounds__(256) void add_ln_kernel(
    const float* __restrict__ X, const float* __restrict__ R,
    const float* __restrict__ g, const float* __restrict__ be,
    float* __restrict__ Y, unsigned short* __restrict__ Ybf, int transpose_in)
{
  __shared__ float red[256];
  __shared__ float stat[2];
  const int n = blockIdx.x;
  int rin = n;
  if (transpose_in) { const int b = n >> 6, q = n & 63; rin = q*BS_ + b; }
  const int t = threadIdx.x;
  const float x = X[(size_t)rin*D_ + t] + R[(size_t)rin*D_ + t];
  red[t] = x; __syncthreads();
  for (int s = 128; s > 0; s >>= 1) { if (t < s) red[t] += red[t + s]; __syncthreads(); }
  if (t == 0) stat[0] = red[0] * (1.0f/D_);
  __syncthreads();
  const float xc = x - stat[0];
  red[t] = xc*xc; __syncthreads();
  for (int s = 128; s > 0; s >>= 1) { if (t < s) red[t] += red[t + s]; __syncthreads(); }
  if (t == 0) stat[1] = rsqrtf(red[0] * (1.0f/D_) + 1e-5f);
  __syncthreads();
  const float y = xc * stat[1] * g[t] + be[t];
  Y[(size_t)n*D_ + t] = y;
  if (Ybf) Ybf[(size_t)n*D_ + t] = f2bf(y);
}

// =====================================================================
extern "C" void kernel_launch(void* const* d_in, const int* in_sizes, int n_in,
                              void* d_out, int out_size, void* d_ws, size_t ws_size,
                              hipStream_t stream)
{
  (void)in_sizes; (void)n_in; (void)out_size; (void)ws_size;

  const float* tgt  = (const float*)d_in[0];
  const float* mem  = (const float*)d_in[1];
  const float* pos  = (const float*)d_in[2];
  const float* qpos = (const float*)d_in[3];
  const float* qse  = (const float*)d_in[4];

  auto P = [&](int i) { return (const float*)d_in[5 + i]; };
  const float *sa_qc_w=P(0),  *sa_qc_b=P(1),  *sa_qp_w=P(2),  *sa_qp_b=P(3);
  const float *sa_kc_w=P(4),  *sa_kc_b=P(5),  *sa_kp_w=P(6),  *sa_kp_b=P(7);
  const float *sa_v_w =P(8),  *sa_v_b =P(9),  *sa_out_w=P(10),*sa_out_b=P(11);
  const float *ca_qc_w=P(12), *ca_qc_b=P(13), *ca_qp_w=P(14), *ca_qp_b=P(15);
  const float *ca_kc_w=P(16), *ca_kc_b=P(17), *ca_kp_w=P(18), *ca_kp_b=P(19);
  const float *ca_v_w =P(20), *ca_v_b =P(21), *ca_qse_w=P(22),*ca_qse_b=P(23);
  const float *ca_out_w=P(24),*ca_out_b=P(25);
  const float *lin1_w=P(26),  *lin1_b=P(27),  *lin2_w=P(28),  *lin2_b=P(29);
  const float *n1_g=P(30), *n1_b=P(31), *n2_g=P(32), *n2_b=P(33), *n3_g=P(34), *n3_b=P(35);

  const int* key_cnt = (const int*)d_in[41];
  const int* ipair   = (const int*)d_in[42];
  const int* ibatch  = (const int*)d_in[43];

  // ---- workspace (16B-aligned bump allocator) ----
  char* base = (char*)d_ws;
  size_t off = 0;
  auto allocf = [&](size_t n) { float* p = (float*)(base + off);
                                off += (n*4 + 15) & ~(size_t)15; return p; };
  auto alloch = [&](size_t n) { unsigned short* p = (unsigned short*)(base + off);
                                off += (n*2 + 15) & ~(size_t)15; return p; };

  const size_t ND  = (size_t)NTOT_ * D_;
  const size_t MD  = (size_t)BS_ * NK_ * D_;
  float* KC  = allocf(MD);  float* KP = allocf(MD);  float* VM = allocf(MD);
  float* OPf = allocf(ND);  float* CQ = allocf(ND);  float* CQSE = allocf(ND);
  float* COPf= allocf(ND);  float* F2f= allocf(ND);
  float* T1  = allocf(ND);  float* T2 = allocf(ND);
  unsigned short* tgt_bf  = alloch(ND);
  unsigned short* qpos_bf = alloch(ND);
  unsigned short* mem_bf  = alloch(MD);
  unsigned short* pos_bf  = alloch(MD);
  unsigned short* Qbf = alloch(ND); unsigned short* Kbf = alloch(ND);
  unsigned short* Vbf = alloch(ND); unsigned short* Obf = alloch(ND);
  unsigned short* T1bf = alloch(ND); unsigned short* T2bf = alloch(ND);
  unsigned short* QPbf = alloch(ND); unsigned short* QSEbf = alloch(ND);
  unsigned short* CObf = alloch(ND);
  unsigned short* F1bf = alloch((size_t)NTOT_ * FF_);
  unsigned short* wt[13];
  for (int i = 0; i < 13; ++i) wt[i] = alloch((size_t)D_ * D_);
  unsigned short* wt_lin1 = alloch((size_t)D_ * FF_);
  unsigned short* wt_lin2 = alloch((size_t)FF_ * D_);

  const dim3 blk(256);
  const dim3 gD(NTOT_/128, D_/64);
  const dim3 gM(BS_*NK_/128, D_/64);
  const dim3 gF(NTOT_/128, FF_/64);

  // ---- conversions ----
  cvt_bf16_kernel<<<(int)(ND/4/256),  blk, 0, stream>>>(tgt,  tgt_bf,  (int)(ND/4));
  cvt_bf16_kernel<<<(int)(ND/4/256),  blk, 0, stream>>>(qpos, qpos_bf, (int)(ND/4));
  cvt_bf16_kernel<<<(int)(MD/4/256),  blk, 0, stream>>>(mem,  mem_bf,  (int)(MD/4));
  cvt_bf16_kernel<<<(int)(MD/4/256),  blk, 0, stream>>>(pos,  pos_bf,  (int)(MD/4));
  const float* wsrc[13] = {sa_qc_w, sa_qp_w, sa_kc_w, sa_kp_w, sa_v_w, sa_out_w,
                           ca_qc_w, ca_qp_w, ca_kc_w, ca_kp_w, ca_v_w, ca_qse_w, ca_out_w};
  for (int i = 0; i < 13; ++i)
    cvt_bf16_t_kernel<<<dim3(D_/32, D_/32), blk, 0, stream>>>(wsrc[i], wt[i], D_, D_);
  cvt_bf16_t_kernel<<<dim3(FF_/32, D_/32), blk, 0, stream>>>(lin1_w, wt_lin1, D_, FF_);
  cvt_bf16_t_kernel<<<dim3(D_/32, FF_/32), blk, 0, stream>>>(lin2_w, wt_lin2, FF_, D_);

  // ---- self attention ----
  gemm_bf16_kernel<<<gD, blk, 0, stream>>>(tgt_bf, wt[0], sa_qc_b, qpos_bf, wt[1], sa_qp_b,
                                           nullptr, Qbf, NTOT_, D_, D_, 0);
  gemm_bf16_kernel<<<gD, blk, 0, stream>>>(tgt_bf, wt[2], sa_kc_b, qpos_bf, wt[3], sa_kp_b,
                                           nullptr, Kbf, NTOT_, D_, D_, 0);
  gemm_bf16_kernel<<<gD, blk, 0, stream>>>(tgt_bf, wt[4], sa_v_b, nullptr, nullptr, nullptr,
                                           nullptr, Vbf, NTOT_, D_, D_, 0);
  self_attn_kernel<<<BS_*H_, 128, 0, stream>>>(Qbf, Kbf, Vbf, Obf);
  gemm_bf16_kernel<<<gD, blk, 0, stream>>>(Obf, wt[5], sa_out_b, nullptr, nullptr, nullptr,
                                           OPf, nullptr, NTOT_, D_, D_, 0);
  add_ln_kernel<<<NTOT_, 256, 0, stream>>>(tgt, OPf, n1_g, n1_b, T1, T1bf, 1);

  // ---- cross attention ----
  reorder_bf16_kernel<<<NTOT_, 256, 0, stream>>>(qpos, QPbf);
  reorder_bf16_kernel<<<NTOT_, 256, 0, stream>>>(qse,  QSEbf);
  gemm_bf16_kernel<<<gD, blk, 0, stream>>>(T1bf, wt[6], ca_qc_b, QPbf, wt[7], ca_qp_b,
                                           CQ, nullptr, NTOT_, D_, D_, 0);
  gemm_bf16_kernel<<<gD, blk, 0, stream>>>(QSEbf, wt[11], ca_qse_b, nullptr, nullptr, nullptr,
                                           CQSE, nullptr, NTOT_, D_, D_, 0);
  gemm_bf16_kernel<<<gM, blk, 0, stream>>>(mem_bf, wt[8], ca_kc_b, nullptr, nullptr, nullptr,
                                           KC, nullptr, BS_*NK_, D_, D_, 0);
  gemm_bf16_kernel<<<gM, blk, 0, stream>>>(pos_bf, wt[9], ca_kp_b, nullptr, nullptr, nullptr,
                                           KP, nullptr, BS_*NK_, D_, D_, 0);
  gemm_bf16_kernel<<<gM, blk, 0, stream>>>(mem_bf, wt[10], ca_v_b, nullptr, nullptr, nullptr,
                                           VM, nullptr, BS_*NK_, D_, D_, 0);
  cross_attn_kernel<<<NTOT_, 256, 0, stream>>>(CQ, CQSE, KC, KP, VM, ipair, ibatch, key_cnt, CObf);
  gemm_bf16_kernel<<<gD, blk, 0, stream>>>(CObf, wt[12], ca_out_b, nullptr, nullptr, nullptr,
                                           COPf, nullptr, NTOT_, D_, D_, 0);
  add_ln_kernel<<<NTOT_, 256, 0, stream>>>(T1, COPf, n2_g, n2_b, T2, T2bf, 0);

  // ---- FFN ----
  gemm_bf16_kernel<<<gF, blk, 0, stream>>>(T2bf, wt_lin1, lin1_b, nullptr, nullptr, nullptr,
                                           nullptr, F1bf, NTOT_, FF_, D_, 1);
  gemm_bf16_kernel<<<gD, blk, 0, stream>>>(F1bf, wt_lin2, lin2_b, nullptr, nullptr, nullptr,
                                           F2f, nullptr, NTOT_, D_, FF_, 0);
  add_ln_kernel<<<NTOT_, 256, 0, stream>>>(T2, F2f, n3_g, n3_b, (float*)d_out, nullptr, 0);
}
